// MultiQueryAttention_88562225643874
// MI455X (gfx1250) — matrix operand order repaired and checked by reading, not compiled, not run on hardware
//
#include <hip/hip_runtime.h>

#define EMBED_DIM 1024
#define NUM_HEADS 16
#define HEAD_DIM  64
#define BB 2
#define TT 2048
#define MROWS (BB * TT)   // 4096

typedef __attribute__((ext_vector_type(16))) __bf16 v16bf;
typedef __attribute__((ext_vector_type(8)))  float  v8f;
typedef __attribute__((ext_vector_type(4)))  float  v4f;
typedef __attribute__((ext_vector_type(4)))  unsigned int v4u;
typedef __attribute__((ext_vector_type(2)))  unsigned int v2u;
typedef __attribute__((ext_vector_type(4)))  unsigned int u32x4;
typedef __attribute__((ext_vector_type(4)))  int i32x4;
typedef __attribute__((ext_vector_type(8)))  int i32x8;

#if defined(__has_builtin)
# if __has_builtin(__builtin_amdgcn_tensor_load_to_lds) && \
     __has_builtin(__builtin_amdgcn_s_wait_tensorcnt)
#  define HAVE_TDM 1
# else
#  define HAVE_TDM 0
# endif
#else
# define HAVE_TDM 0
#endif

union Frag16 {
    v16bf v;
    unsigned short u[16];
    v4u q[2];
};

// f32 -> bf16 via the native converter (RNE on gfx1250 bf16 VALU)
static __device__ __forceinline__ unsigned short f2bf(float f) {
    __bf16 b = (__bf16)f;
    return __builtin_bit_cast(unsigned short, b);
}
static __device__ __forceinline__ unsigned int f2bf2(float lo, float hi) {
    return (unsigned int)f2bf(lo) | ((unsigned int)f2bf(hi) << 16);
}

static __device__ __forceinline__ v8f wmma_bf16(const Frag16& a, const Frag16& b, v8f c) {
    return __builtin_amdgcn_wmma_f32_16x16x32_bf16(false, a.v, false, b.v,
                                                   (short)0, c, false, false);
}

#if HAVE_TDM
// LDS byte offset of a __shared__ object (generic -> addrspace(3) -> int)
static __device__ __forceinline__ unsigned lds_byte_off(const void* p) {
    return (unsigned)(unsigned long long)
        (__attribute__((address_space(3))) const void*)p;
}
#endif

// ---------------------------------------------------------------------------
// Tiled GEMM:  Out[M,N] = (A[M,K] @ W[K,N] + bias) * scale
// block = 256 threads = 8 waves; block tile = 32(M) x 128(N); K step = 32.
// Each wave owns one 16-col N-tile and two stacked 16x16 M-tiles.
// W is staged TRANSPOSED in LDS so B-fragments load as 2x ds_load_b128.
// ---------------------------------------------------------------------------
template <bool A_BF16, bool OUT_BF16>
__global__ __launch_bounds__(256)
void gemm_bias_kernel(const void* __restrict__ Av, const float* __restrict__ W,
                      const float* __restrict__ bias, void* __restrict__ Ov,
                      int M, int N, int K, float scale)
{
    __shared__ __align__(16) unsigned short As[32 * 32];    // [32 m][32 k] bf16
    __shared__ __align__(16) unsigned short Wt[128 * 32];   // [128 n][32 k] bf16 (transposed)

    const int tid  = threadIdx.x;
    const int wave = tid >> 5;
    const int lane = tid & 31;
    const int m0   = blockIdx.y * 32;
    const int n0   = blockIdx.x * 128;

    const int t  = lane & 15;                 // fragment row / column index
    const int kh = lane >> 4;                 // K-half of the wave
    const int cl   = wave * 16 + t;           // local output column
    const int ncol = n0 + cl;                 // global output column
    const bool wave_active = (n0 + wave * 16) < N;   // uniform per wave

    // staging coordinates
    const int sa_r = tid >> 3;                // A: row 0..31
    const int sa_c = (tid & 7) * 4;           // A: 4 consecutive k
    const int sw_c = tid & 127;               // W: column
    const int sw_r0 = (tid >> 7) * 16;        // W: k rows r0..r0+15

    v8f acc0 = {}, acc1 = {};

    for (int kb = 0; kb < K; kb += 32) {
        // ---- stage A tile (32x32), 4 halves per thread, packed b64 store ----
        {
            v2u pk;
            if (A_BF16) {
                const unsigned short* A = (const unsigned short*)Av;
                pk = *(const v2u*)(A + (size_t)(m0 + sa_r) * K + kb + sa_c);
            } else {
                const float* A = (const float*)Av;
                v4f f = *(const v4f*)(A + (size_t)(m0 + sa_r) * K + kb + sa_c);
                pk[0] = f2bf2(f[0], f[1]);
                pk[1] = f2bf2(f[2], f[3]);
            }
            *(v2u*)&As[sa_r * 32 + sa_c] = pk;
        }
        // ---- stage W tile transposed (Wt[n][k]), packed b32 stores ----
        if (n0 + sw_c < N) {
            #pragma unroll
            for (int i = 0; i < 16; i += 2) {
                float f0 = W[(size_t)(kb + sw_r0 + i)     * N + n0 + sw_c];
                float f1 = W[(size_t)(kb + sw_r0 + i + 1) * N + n0 + sw_c];
                *(unsigned int*)&Wt[sw_c * 32 + sw_r0 + i] = f2bf2(f0, f1);
            }
        }
        __syncthreads();

        if (wave_active) {
            Frag16 a0, a1, b;
            a0.q[0] = *(const v4u*)&As[t * 32 + kh * 8];
            a0.q[1] = *(const v4u*)&As[t * 32 + 16 + kh * 8];
            a1.q[0] = *(const v4u*)&As[(16 + t) * 32 + kh * 8];
            a1.q[1] = *(const v4u*)&As[(16 + t) * 32 + 16 + kh * 8];
            b.q[0]  = *(const v4u*)&Wt[cl * 32 + kh * 8];
            b.q[1]  = *(const v4u*)&Wt[cl * 32 + 16 + kh * 8];
            acc0 = wmma_bf16(a0, b, acc0);
            acc1 = wmma_bf16(a1, b, acc1);
        }
        __syncthreads();
    }

    if (wave_active && ncol < N) {
        const float bval = bias[ncol];
        #pragma unroll
        for (int r = 0; r < 8; ++r) {
            const int row0 = m0 + kh * 8 + r;        // C layout: VGPR r -> M = r + kh*8
            const int row1 = row0 + 16;
            const float v0 = (acc0[r] + bval) * scale;
            const float v1 = (acc1[r] + bval) * scale;
            if (OUT_BF16) {
                ((unsigned short*)Ov)[(size_t)row0 * N + ncol] = f2bf(v0);
                ((unsigned short*)Ov)[(size_t)row1 * N + ncol] = f2bf(v1);
            } else {
                ((float*)Ov)[(size_t)row0 * N + ncol] = v0;
                ((float*)Ov)[(size_t)row1 * N + ncol] = v1;
            }
        }
    }
}

// ---------------------------------------------------------------------------
// MQA flash attention.  One wave32 per (batch, head, 16-row Q tile).
// Q tile pulled into LDS by the Tensor Data Mover (TDM).
// V chunk staged transposed (v_perm_b32 packing) so P.V B-fragments are
// contiguous ds_load_b128.
// ---------------------------------------------------------------------------
__global__ __launch_bounds__(32)
void mqa_flash_kernel(const unsigned short* __restrict__ Q,
                      const unsigned short* __restrict__ Kb,
                      const unsigned short* __restrict__ Vb,
                      unsigned short* __restrict__ Ob)
{
    __shared__ __align__(16) unsigned short Ql[16 * 64];   // Q tile bf16 (row-major)
    __shared__ __align__(16) unsigned short Pl[16 * 32];   // P tile bf16
    __shared__ __align__(16) unsigned short Vt[64 * 32];   // V chunk bf16, [d][s] transposed

    const int lane = threadIdx.x;
    const int qt   = blockIdx.x;      // q tile (0..127)
    const int h    = blockIdx.y;      // head  (0..15)
    const int bz   = blockIdx.z;      // batch (0..1)
    const int t    = lane & 15;
    const int kh   = lane >> 4;

    Frag16 qa0, qa1;
#if HAVE_TDM
    {
        // --- Tensor DMA: 16x64 bf16 tile of Q[(bz*TT+qt*16) .., h*64 ..] -> Ql ---
        const unsigned long long ga = (unsigned long long)(size_t)
            (Q + (size_t)(bz * TT + qt * 16) * EMBED_DIM + h * HEAD_DIM);
        u32x4 g0;
        g0[0] = 1u;                                        // count=1, user descriptor
        g0[1] = lds_byte_off(Ql);                          // lds_addr
        g0[2] = (unsigned)ga;                              // global_addr[31:0]
        g0[3] = (unsigned)(ga >> 32) | (2u << 30);         // global_addr[56:32] | type=2
        i32x8 g1;
        g1[0] = (int)(1u << 16);                           // data_size = 2 bytes
        g1[1] = (int)((unsigned)EMBED_DIM << 16);          // tensor_dim0[15:0] = 1024
        g1[2] = (int)(((unsigned)MROWS & 0xFFFFu) << 16);  // dim0[31:16]=0 | dim1[15:0]=4096
        g1[3] = (int)((unsigned)HEAD_DIM << 16);           // dim1[31:16]=0 | tile_dim0=64
        g1[4] = 16;                                        // tile_dim1 = 16 rows (tile_dim2=0)
        g1[5] = EMBED_DIM;                                 // tensor_dim0_stride = 1024
        g1[6] = 0;
        g1[7] = 0;
        const i32x4 gz4 = {0, 0, 0, 0};
        const i32x8 gz8 = {0, 0, 0, 0, 0, 0, 0, 0};
        __builtin_amdgcn_tensor_load_to_lds(g0, g1, gz4, gz4, gz8, 0);
        __builtin_amdgcn_s_wait_tensorcnt(0);
        // A-fragments out of LDS (row t, 64 halves per row)
        qa0.q[0] = *(const v4u*)&Ql[t * 64 + kh * 8];
        qa0.q[1] = *(const v4u*)&Ql[t * 64 + 16 + kh * 8];
        qa1.q[0] = *(const v4u*)&Ql[t * 64 + 32 + kh * 8];
        qa1.q[1] = *(const v4u*)&Ql[t * 64 + 48 + kh * 8];
    }
#else
    {
        const unsigned short* qr =
            Q + (size_t)(bz * TT + qt * 16 + t) * EMBED_DIM + h * HEAD_DIM;
        qa0.q[0] = *(const v4u*)&qr[kh * 8];
        qa0.q[1] = *(const v4u*)&qr[16 + kh * 8];
        qa1.q[0] = *(const v4u*)&qr[32 + kh * 8];
        qa1.q[1] = *(const v4u*)&qr[48 + kh * 8];
        (void)Ql;
    }
#endif

    const unsigned short* Kbase = Kb + (size_t)bz * TT * HEAD_DIM;
    const unsigned short* Vbase = Vb + (size_t)bz * TT * HEAD_DIM;

    float m[8], l[8];
    #pragma unroll
    for (int r = 0; r < 8; ++r) { m[r] = -3.0e38f; l[r] = 0.0f; }
    v8f o[4] = {};

    for (int s0 = 0; s0 < TT; s0 += 32) {
        // ---- scores: S_tile = Q(16x64) . K^T(64x32); K^T fragments are
        //      K-contiguous in row-major K, so plain b128 global loads ----
        v8f c0 = {}, c1 = {};
        {
            const int sc0 = s0 + t;
            const int sc1 = s0 + 16 + t;
            const int kb0 = kh * 8;
            Frag16 kf;
            kf.q[0] = *(const v4u*)&Kbase[(size_t)sc0 * 64 + kb0];
            kf.q[1] = *(const v4u*)&Kbase[(size_t)sc0 * 64 + kb0 + 16];
            c0 = wmma_bf16(qa0, kf, c0);
            kf.q[0] = *(const v4u*)&Kbase[(size_t)sc0 * 64 + 32 + kb0];
            kf.q[1] = *(const v4u*)&Kbase[(size_t)sc0 * 64 + 32 + kb0 + 16];
            c0 = wmma_bf16(qa1, kf, c0);
            kf.q[0] = *(const v4u*)&Kbase[(size_t)sc1 * 64 + kb0];
            kf.q[1] = *(const v4u*)&Kbase[(size_t)sc1 * 64 + kb0 + 16];
            c1 = wmma_bf16(qa0, kf, c1);
            kf.q[0] = *(const v4u*)&Kbase[(size_t)sc1 * 64 + 32 + kb0];
            kf.q[1] = *(const v4u*)&Kbase[(size_t)sc1 * 64 + 32 + kb0 + 16];
            c1 = wmma_bf16(qa1, kf, c1);
        }

        // ---- stage V chunk transposed: Vt[d][s] = V[s0+s][d]
        //      lane handles s pair {2t, 2t+1} and d range kh*32..+31;
        //      one v_perm_b32 per packed b32 store ----
        {
            const int dbase = kh * 32;
            union { v4u v[4]; unsigned int w[16]; } rowA, rowB;
            const unsigned short* vr0 = Vbase + (size_t)(s0 + 2 * t) * 64 + dbase;
            const unsigned short* vr1 = vr0 + 64;
            #pragma unroll
            for (int j = 0; j < 4; ++j) {
                rowA.v[j] = *(const v4u*)(vr0 + j * 8);
                rowB.v[j] = *(const v4u*)(vr1 + j * 8);
            }
            #pragma unroll
            for (int j = 0; j < 16; ++j) {
                const unsigned wa = rowA.w[j], wb = rowB.w[j];
                *(unsigned int*)&Vt[(dbase + 2 * j)     * 32 + 2 * t] =
                    __builtin_amdgcn_perm(wb, wa, 0x05040100u);   // lo16(A) | lo16(B)<<16
                *(unsigned int*)&Vt[(dbase + 2 * j + 1) * 32 + 2 * t] =
                    __builtin_amdgcn_perm(wb, wa, 0x07060302u);   // hi16(A) | hi16(B)<<16
            }
        }

        // ---- online softmax (rows live across 16 lanes; shfl width 16) ----
        #pragma unroll
        for (int r = 0; r < 8; ++r) {
            float sm = fmaxf(c0[r], c1[r]);
            sm = fmaxf(sm, __shfl_xor(sm, 1, 16));
            sm = fmaxf(sm, __shfl_xor(sm, 2, 16));
            sm = fmaxf(sm, __shfl_xor(sm, 4, 16));
            sm = fmaxf(sm, __shfl_xor(sm, 8, 16));
            const float mn  = fmaxf(m[r], sm);
            const float fac = __expf(m[r] - mn);
            const float p0  = __expf(c0[r] - mn);
            const float p1  = __expf(c1[r] - mn);
            float rs = p0 + p1;
            rs += __shfl_xor(rs, 1, 16);
            rs += __shfl_xor(rs, 2, 16);
            rs += __shfl_xor(rs, 4, 16);
            rs += __shfl_xor(rs, 8, 16);
            l[r] = l[r] * fac + rs;
            m[r] = mn;
            const int trow = r + kh * 8;            // C layout row
            Pl[trow * 32 + t]      = f2bf(p0);
            Pl[trow * 32 + 16 + t] = f2bf(p1);
            o[0][r] *= fac; o[1][r] *= fac; o[2][r] *= fac; o[3][r] *= fac;
        }

        // ---- O += P(16x32) . V(32x64); all fragments contiguous b128 ----
        Frag16 pa;
        pa.q[0] = *(const v4u*)&Pl[t * 32 + kh * 8];
        pa.q[1] = *(const v4u*)&Pl[t * 32 + 16 + kh * 8];
        #pragma unroll
        for (int nt = 0; nt < 4; ++nt) {
            Frag16 vf;
            const int dc = nt * 16 + t;
            vf.q[0] = *(const v4u*)&Vt[dc * 32 + kh * 8];
            vf.q[1] = *(const v4u*)&Vt[dc * 32 + 16 + kh * 8];
            o[nt] = wmma_bf16(pa, vf, o[nt]);
        }
    }

    // ---- epilogue: normalize and store bf16 [B*T, E] ----
    #pragma unroll
    for (int r = 0; r < 8; ++r) {
        const float inv = 1.0f / l[r];
        const size_t row = (size_t)(bz * TT + qt * 16 + kh * 8 + r);
        #pragma unroll
        for (int nt = 0; nt < 4; ++nt) {
            const int col = h * HEAD_DIM + nt * 16 + t;
            Ob[row * EMBED_DIM + col] = f2bf(o[nt][r] * inv);
        }
    }
}

// ---------------------------------------------------------------------------
extern "C" void kernel_launch(void* const* d_in, const int* in_sizes, int n_in,
                              void* d_out, int out_size, void* d_ws, size_t ws_size,
                              hipStream_t stream) {
    const float* query = (const float*)d_in[0];
    const float* key_  = (const float*)d_in[1];
    const float* value = (const float*)d_in[2];
    const float* Wq    = (const float*)d_in[3];
    const float* bq    = (const float*)d_in[4];
    const float* Wk    = (const float*)d_in[5];
    const float* bk    = (const float*)d_in[6];
    const float* Wv    = (const float*)d_in[7];
    const float* bv    = (const float*)d_in[8];
    const float* Wo    = (const float*)d_in[9];
    const float* bo    = (const float*)d_in[10];
    float* out = (float*)d_out;

    char* ws = (char*)d_ws;
    unsigned short* qbf = (unsigned short*)(ws);                          // 8 MB
    unsigned short* kbf = (unsigned short*)(ws + 8388608);                // 512 KB
    unsigned short* vbf = (unsigned short*)(ws + 8388608 + 524288);       // 512 KB
    unsigned short* abf = (unsigned short*)(ws + 8388608 + 2 * 524288);   // 8 MB

    const float qscale = 0.125f;   // HEAD_DIM^-0.5

    // Q/K/V projections (f32 in, bf16 out; softmax scale folded into Q)
    gemm_bias_kernel<false, true><<<dim3(EMBED_DIM / 128, MROWS / 32), 256, 0, stream>>>(
        query, Wq, bq, qbf, MROWS, EMBED_DIM, EMBED_DIM, qscale);
    gemm_bias_kernel<false, true><<<dim3(1, MROWS / 32), 256, 0, stream>>>(
        key_, Wk, bk, kbf, MROWS, HEAD_DIM, EMBED_DIM, 1.0f);
    gemm_bias_kernel<false, true><<<dim3(1, MROWS / 32), 256, 0, stream>>>(
        value, Wv, bv, vbf, MROWS, HEAD_DIM, EMBED_DIM, 1.0f);

    // flash attention: one wave per (b, h, 16-row q tile)
    mqa_flash_kernel<<<dim3(TT / 16, NUM_HEADS, BB), 32, 0, stream>>>(qbf, kbf, vbf, abf);

    // output projection (bf16 in, f32 out)
    gemm_bias_kernel<true, false><<<dim3(EMBED_DIM / 128, MROWS / 32), 256, 0, stream>>>(
        abf, Wo, bo, out, MROWS, EMBED_DIM, EMBED_DIM, 1.0f);
}
